// CFormer_62646392980129
// MI455X (gfx1250) — matrix operand order repaired and checked by reading, not compiled, not run on hardware
//
#include <hip/hip_runtime.h>
#include <hip/hip_bf16.h>
#include <cmath>

// ---------------------------------------------------------------------------
// CFormer forward for MI455X (gfx1250, wave32):
//   - bf16 WMMA (v_wmma_f32_16x16x32_bf16) for all GEMMs / attention
//   - Tensor Data Mover (tensor_load_to_lds + s_wait_tensorcnt) for tile DMA,
//     double-buffered and pipelined one step ahead of the WMMA compute
// ---------------------------------------------------------------------------

typedef __bf16 v16bf __attribute__((ext_vector_type(16)));
typedef __bf16 v8bf  __attribute__((ext_vector_type(8)));
typedef float  v8f   __attribute__((ext_vector_type(8)));
typedef unsigned int u32x4 __attribute__((ext_vector_type(4)));
typedef int          i32x4 __attribute__((ext_vector_type(4)));
typedef int          i32x8 __attribute__((ext_vector_type(8)));

#ifndef __has_builtin
#define __has_builtin(x) 0
#endif
#if __has_builtin(__builtin_amdgcn_tensor_load_to_lds) && \
    __has_builtin(__builtin_amdgcn_s_wait_tensorcnt)
#define USE_TDM 1
#else
#define USE_TDM 0
#endif

static constexpr int Bn_   = 4;
static constexpr int T_    = 1500;
static constexpr int D_    = 1280;
static constexpr int FFN_  = 5120;
static constexpr int H_    = 8;
static constexpr int HD_   = 160;
static constexpr int MT_   = 150;     // MAX_TOKENS
static constexpr int OUT_  = 4096;
static constexpr int VOC_  = 32000;

static __device__ __forceinline__ v8f zero8f() {
    v8f r;
#pragma unroll
    for (int i = 0; i < 8; ++i) r[i] = 0.0f;
    return r;
}
static __device__ __forceinline__ v8bf zero8bf() {
    v8bf r;
#pragma unroll
    for (int i = 0; i < 8; ++i) r[i] = (__bf16)0.0f;
    return r;
}
static __device__ __forceinline__ v16bf cat16(v8bf lo, v8bf hi) {
    v16bf r;
#pragma unroll
    for (int e = 0; e < 8; ++e) { r[e] = lo[e]; r[e + 8] = hi[e]; }
    return r;
}
// A-matrix fragment (16x32 bf16): lane(half,m) holds K=16h..16h+15 contiguous
// (same k order as the B fragment).
static __device__ __forceinline__ v16bf ld_avec(const __bf16* row, int half) {
    v8bf lo = *(const v8bf*)(row + 16 * half);
    v8bf hi = *(const v8bf*)(row + 16 * half + 8);
    return cat16(lo, hi);
}
// B-matrix fragment (32x16 bf16): lane(half,n) holds K=16h..16h+15 contiguous.
static __device__ __forceinline__ v16bf ld_bvec(const __bf16* row, int half) {
    v8bf lo = *(const v8bf*)(row + 16 * half);
    v8bf hi = *(const v8bf*)(row + 16 * half + 8);
    return cat16(lo, hi);
}
static __device__ __forceinline__ v8f wmma_bf16(v16bf a, v16bf b, v8f c) {
    return __builtin_amdgcn_wmma_f32_16x16x32_bf16(false, a, false, b, (short)0, c, false, false);
}
static __device__ __forceinline__ float gelu_exact(float x) {
    return 0.5f * x * (1.0f + erff(x * 0.70710678118654752f));
}

#if USE_TDM
// ---------------------------------------------------------------------------
// Tensor Data Mover: DMA a 2D bf16 tile [tile_rows x tile_cols] from global
// (row stride = stride_elems) into LDS at byte offset lds_off. OOB rows/cols
// (beyond valid_*) are zero-filled by the TDM. pad_on inserts 4 dwords of LDS
// padding every 16 dwords -> LDS row stride 40 elements for a 32-elem row.
// D# layout per CDNA5 ISA ch. 8 (group0 128b, group1 256b; 2D => groups 2/3
// unused / zero).
// ---------------------------------------------------------------------------
static __device__ __forceinline__ void tdm_load_tile_bf16(
    unsigned lds_off, const __bf16* gptr, long stride_elems,
    unsigned valid_cols, unsigned valid_rows,
    unsigned tile_cols, unsigned tile_rows, int pad_on)
{
    const unsigned long long ga = (unsigned long long)(const void*)gptr;
    u32x4 g0;
    g0[0] = 1u;                                     // count=1, user descriptor
    g0[1] = lds_off;                                // lds_addr
    g0[2] = (unsigned)ga;                           // global_addr[31:0]
    g0[3] = (unsigned)((ga >> 32) & 0x01ffffffULL)  // global_addr[56:32]
          | (2u << 30);                             // type = 2 ("image")
    i32x8 g1;
    unsigned w0 = (1u << 16);                       // data_size = 2 bytes
    if (pad_on) w0 |= (1u << 20)                    // pad_enable
              | (3u << 22)                          // pad_interval: 16 dwords
              | (3u << 25);                         // pad_amount:   4 dwords
    g1[0] = (int)w0;
    g1[1] = (int)((valid_cols & 0xffffu) << 16);    // tensor_dim0[15:0]
    g1[2] = (int)((valid_cols >> 16) | ((valid_rows & 0xffffu) << 16));
    g1[3] = (int)((valid_rows >> 16) | (tile_cols << 16));  // tile_dim0
    g1[4] = (int)(tile_rows & 0xffffu);             // tile_dim1 (tile_dim2=0)
    const unsigned long long s0 = (unsigned long long)stride_elems;
    g1[5] = (int)(unsigned)s0;                      // tensor_dim0_stride[31:0]
    g1[6] = (int)(unsigned)((s0 >> 32) & 0xffffu);  // stride[47:32]; dim1_stride=0
    g1[7] = 0;
    i32x4 z4; z4[0] = 0; z4[1] = 0; z4[2] = 0; z4[3] = 0;
#if defined(__clang_major__) && __clang_major__ >= 23
    i32x8 z8;
#pragma unroll
    for (int i = 0; i < 8; ++i) z8[i] = 0;
    __builtin_amdgcn_tensor_load_to_lds(g0, g1, z4, z4, z8, 0);
#else
    __builtin_amdgcn_tensor_load_to_lds(g0, g1, z4, z4, 0);
#endif
}
#endif  // USE_TDM

// ---------------------------------------------------------------------------
// WMMA GEMM: C[M,N] = act(((A[M,K] * B^T) + bias) * alpha) (+ resid)
//   Fast path (bf16 A and B, B stored [N,K]): TDM double-buffered pipeline.
//   Fallback (f32 inputs / [K,N] B): guarded scalar staging with conversion.
//   block = 128 threads = 4 waves; 64x64 tile; wave computes 32x32 (2x2 WMMA)
// ---------------------------------------------------------------------------
#define GLD 40   // LDS row stride (elements); 80B rows keep 16B alignment

static __device__ __forceinline__ void gemm_step(
    const __bf16* sA, const __bf16* sB, int wm, int wn, int ln, int half,
    v8f (&acc)[2][2])
{
    const v16bf a0 = ld_avec(&sA[(wm + ln) * GLD], half);
    const v16bf a1 = ld_avec(&sA[(wm + 16 + ln) * GLD], half);
    const v16bf b0 = ld_bvec(&sB[(wn + ln) * GLD], half);
    const v16bf b1 = ld_bvec(&sB[(wn + 16 + ln) * GLD], half);
    acc[0][0] = wmma_bf16(a0, b0, acc[0][0]);
    acc[0][1] = wmma_bf16(a0, b1, acc[0][1]);
    acc[1][0] = wmma_bf16(a1, b0, acc[1][0]);
    acc[1][1] = wmma_bf16(a1, b1, acc[1][1]);
}

__global__ __launch_bounds__(128)
void k_gemm(const void* __restrict__ Ap, int a_f32, long lda, long strideA,
            const void* __restrict__ Bp, int b_f32, int b_kn, long ldb, long strideB,
            const float* __restrict__ bias,
            const float* __restrict__ resid, long ldr, long strideR,
            float* __restrict__ Cf, __bf16* __restrict__ Cb, long ldc, long strideC,
            int M, int N, int K, float alpha, int act)
{
    __shared__ __align__(16) __bf16 sA[2][64 * GLD];
    __shared__ __align__(16) __bf16 sB[2][64 * GLD];

    const int tid  = threadIdx.x;
    const int wave = tid >> 5, lane = tid & 31;
    const int half = lane >> 4, ln = lane & 15;
    const int wm = (wave >> 1) * 32, wn = (wave & 1) * 32;
    const int m0 = blockIdx.y * 64, n0 = blockIdx.x * 64;
    const int bz = blockIdx.z;

    const char* Ab = (const char*)Ap;
    const char* Bb = (const char*)Bp;
    const size_t esA = a_f32 ? 4u : 2u, esB = b_f32 ? 4u : 2u;
    Ab += (size_t)bz * (size_t)strideA * esA;
    Bb += (size_t)bz * (size_t)strideB * esB;

    v8f acc[2][2];
#pragma unroll
    for (int i = 0; i < 2; ++i)
#pragma unroll
        for (int j = 0; j < 2; ++j) acc[i][j] = zero8f();

#if USE_TDM
    if (!a_f32 && !b_f32 && !b_kn) {
        // ---- TDM pipelined path: DMA next tile while computing current ----
        const __bf16* Abf = (const __bf16*)Ab;
        const __bf16* Bbf = (const __bf16*)Bb;
        const int nk = (K + 31) >> 5;
        auto issue = [&](int ki, int buf) {
            const int k0 = ki << 5;
            tdm_load_tile_bf16((unsigned)(size_t)&sA[buf][0],
                               Abf + (size_t)m0 * lda + k0, lda,
                               (unsigned)(K - k0), (unsigned)(M - m0), 32u, 64u, 1);
            tdm_load_tile_bf16((unsigned)(size_t)&sB[buf][0],
                               Bbf + (size_t)n0 * ldb + k0, ldb,
                               (unsigned)(K - k0), (unsigned)(N - n0), 32u, 64u, 1);
        };
        if (wave == 0) issue(0, 0);
        for (int ki = 0; ki < nk; ++ki) {
            const int cur = ki & 1;
            if (wave == 0) {
                if (ki + 1 < nk) {
                    issue(ki + 1, cur ^ 1);
                    __builtin_amdgcn_s_wait_tensorcnt(2);   // pair ki done
                } else {
                    __builtin_amdgcn_s_wait_tensorcnt(0);   // last pair done
                }
            }
            __syncthreads();                     // tile ki visible to all waves
            gemm_step(&sA[cur][0], &sB[cur][0], wm, wn, ln, half, acc);
            __syncthreads();                     // buffer free before reissue
        }
    } else
#endif
    {
        // ---- fallback: guarded scalar staging (f32 convert / [K,N] B) ----
        for (int k0 = 0; k0 < K; k0 += 32) {
            for (int i = tid; i < 64 * 32; i += 128) {
                const int r = i >> 5, c = i & 31;
                const int gm = m0 + r, gk = k0 + c;
                __bf16 val = (__bf16)0.0f;
                if (gm < M && gk < K) {
                    val = a_f32 ? (__bf16)(((const float*)Ab)[(size_t)gm * lda + gk])
                                : ((const __bf16*)Ab)[(size_t)gm * lda + gk];
                }
                sA[0][r * GLD + c] = val;
            }
            if (!b_kn) {
                for (int i = tid; i < 64 * 32; i += 128) {
                    const int r = i >> 5, c = i & 31;
                    const int gn = n0 + r, gk = k0 + c;
                    __bf16 val = (__bf16)0.0f;
                    if (gn < N && gk < K) {
                        val = b_f32 ? (__bf16)(((const float*)Bb)[(size_t)gn * ldb + gk])
                                    : ((const __bf16*)Bb)[(size_t)gn * ldb + gk];
                    }
                    sB[0][r * GLD + c] = val;
                }
            } else {   // B stored [K,N]: transpose while staging
                for (int i = tid; i < 64 * 32; i += 128) {
                    const int r = i & 63, c = i >> 6;
                    const int gn = n0 + r, gk = k0 + c;
                    __bf16 val = (__bf16)0.0f;
                    if (gn < N && gk < K) {
                        val = b_f32 ? (__bf16)(((const float*)Bb)[(size_t)gk * ldb + gn])
                                    : ((const __bf16*)Bb)[(size_t)gk * ldb + gn];
                    }
                    sB[0][r * GLD + c] = val;
                }
            }
            __syncthreads();
            if (k0 + 32 < K) {
                const int r = tid & 63;
                if (m0 + r < M)
                    __builtin_prefetch(Ab + ((size_t)(m0 + r) * lda + k0 + 32) * esA, 0, 1);
            }
            gemm_step(&sA[0][0], &sB[0][0], wm, wn, ln, half, acc);
            __syncthreads();
        }
    }

    // ---- epilogue (C layout: M = v + 8*half, N = ln) ----
#pragma unroll
    for (int ti = 0; ti < 2; ++ti)
#pragma unroll
        for (int tj = 0; tj < 2; ++tj)
#pragma unroll
            for (int v = 0; v < 8; ++v) {
                const int gm = m0 + wm + ti * 16 + v + 8 * half;
                const int gn = n0 + wn + tj * 16 + ln;
                if (gm < M && gn < N) {
                    float x = acc[ti][tj][v];
                    if (bias) x += bias[gn];
                    x *= alpha;                 // q-proj scales after bias
                    if (act == 1) x = gelu_exact(x);
                    if (resid) x += resid[(size_t)bz * strideR + (size_t)gm * ldr + gn];
                    const size_t o = (size_t)bz * strideC + (size_t)gm * ldc + gn;
                    if (Cf) Cf[o] = x;
                    if (Cb) Cb[o] = (__bf16)x;
                }
            }
}

// ---------------------------------------------------------------------------
// LayerNorm: per-row mean/var over D, output bf16
// ---------------------------------------------------------------------------
__global__ __launch_bounds__(256)
void k_layernorm(const float* __restrict__ X, const float* __restrict__ S,
                 const float* __restrict__ Bv, __bf16* __restrict__ O, int Dm)
{
    __shared__ float red[256];
    const int row = blockIdx.x, tid = threadIdx.x;
    const float* x = X + (size_t)row * Dm;

    float s1 = 0.0f;
    for (int i = tid; i < Dm; i += 256) s1 += x[i];
    red[tid] = s1; __syncthreads();
    for (int off = 128; off; off >>= 1) { if (tid < off) red[tid] += red[tid + off]; __syncthreads(); }
    const float mean = red[0] / (float)Dm;
    __syncthreads();

    float s2 = 0.0f;
    for (int i = tid; i < Dm; i += 256) { const float d = x[i] - mean; s2 += d * d; }
    red[tid] = s2; __syncthreads();
    for (int off = 128; off; off >>= 1) { if (tid < off) red[tid] += red[tid + off]; __syncthreads(); }
    const float rstd = rsqrtf(red[0] / (float)Dm + 1e-5f);

    __bf16* o = O + (size_t)row * Dm;
    for (int i = tid; i < Dm; i += 256)
        o[i] = (__bf16)((x[i] - mean) * rstd * S[i] + Bv[i]);
}

// ---------------------------------------------------------------------------
// Flash attention: one wave per (16-query tile, head, batch). WMMA for QK^T
// and PV; online softmax via wave32 half-lane shuffles; P relayout via LDS;
// V tiles DMA'd by TDM (double-buffered, one kv-step ahead).
// q is pre-scaled by HD^-0.5 in its projection.
// ---------------------------------------------------------------------------
#define ALD 40

__global__ __launch_bounds__(32)
void k_attn(const __bf16* __restrict__ Qb, const __bf16* __restrict__ Kb,
            const __bf16* __restrict__ Vb, __bf16* __restrict__ Ob,
            int L, int Dm, const int* __restrict__ ntok)
{
    __shared__ __align__(16) __bf16 pT[16 * ALD];       // P tile (A-layout src)
    __shared__ __align__(16) __bf16 vS[2][32 * HD_];    // V tiles, row-major

    const int qt = blockIdx.x, h = blockIdx.y, b = blockIdx.z;
    const int lane = threadIdx.x;
    const int half = lane >> 4, ln = lane & 15;
    const int vlen = ntok ? max(ntok[b], 1) : L;
    const size_t base = (size_t)b * L * Dm + (size_t)h * HD_;

    // Q tile -> 5 A fragments (K dim = HD = 160 = 5 * 32)
    v16bf aq[5];
    const int qr = qt * 16 + ln;
#pragma unroll
    for (int kk = 0; kk < 5; ++kk) {
        v8bf lo = zero8bf(), hi = zero8bf();
        if (qr < L) {
            const __bf16* qp = Qb + base + (size_t)qr * Dm + kk * 32;
            lo = *(const v8bf*)(qp + 16 * half);
            hi = *(const v8bf*)(qp + 16 * half + 8);
        }
        aq[kk] = cat16(lo, hi);
    }

    v8f acc[10];
#pragma unroll
    for (int d = 0; d < 10; ++d) acc[d] = zero8f();
    float rmax[8], rsum[8];
#pragma unroll
    for (int v = 0; v < 8; ++v) { rmax[v] = -3.0e38f; rsum[v] = 0.0f; }

    const int nsteps = (L + 31) >> 5;
#if USE_TDM
    auto issueV = [&](int j, int buf) {
        const int kv0 = j << 5;
        tdm_load_tile_bf16((unsigned)(size_t)&vS[buf][0],
                           Vb + base + (size_t)kv0 * Dm, Dm,
                           (unsigned)HD_, (unsigned)(L - kv0),
                           (unsigned)HD_, 32u, 0);
    };
    issueV(0, 0);
#endif
    for (int j = 0; j < nsteps; ++j) {
        const int kv0 = j << 5;
        const int cur = j & 1;

        // ---- scores S = Q K^T for 32 kv columns ----
        v8f sc[2] = { zero8f(), zero8f() };
#pragma unroll
        for (int t2 = 0; t2 < 2; ++t2) {
            const int kvr = kv0 + t2 * 16 + ln;
            const bool ok = (kvr < L);
            const __bf16* kp = Kb + base + (size_t)kvr * Dm;
#pragma unroll
            for (int kk = 0; kk < 5; ++kk) {
                v8bf lo = zero8bf(), hi = zero8bf();
                if (ok) {
                    lo = *(const v8bf*)(kp + kk * 32 + 16 * half);
                    hi = *(const v8bf*)(kp + kk * 32 + 16 * half + 8);
                }
                sc[t2] = wmma_bf16(aq[kk], cat16(lo, hi), sc[t2]);
            }
        }
        // mask invalid / padded columns
#pragma unroll
        for (int t2 = 0; t2 < 2; ++t2) {
            const int col = kv0 + t2 * 16 + ln;
            if (col >= vlen) {
#pragma unroll
                for (int v = 0; v < 8; ++v) sc[t2][v] = -1.0e30f;
            }
        }

        // ---- online softmax (row reductions across 16-lane halves) ----
        float corr[8];
#pragma unroll
        for (int v = 0; v < 8; ++v) {
            float mx = fmaxf(sc[0][v], sc[1][v]);
            mx = fmaxf(mx, __shfl_xor(mx, 1, 32));
            mx = fmaxf(mx, __shfl_xor(mx, 2, 32));
            mx = fmaxf(mx, __shfl_xor(mx, 4, 32));
            mx = fmaxf(mx, __shfl_xor(mx, 8, 32));
            const float nm = fmaxf(rmax[v], mx);
            const float c  = __expf(rmax[v] - nm);
            rmax[v] = nm; corr[v] = c;
            const float p0 = __expf(sc[0][v] - nm);
            const float p1 = __expf(sc[1][v] - nm);
            sc[0][v] = p0; sc[1][v] = p1;
            float s = p0 + p1;
            s += __shfl_xor(s, 1, 32);
            s += __shfl_xor(s, 2, 32);
            s += __shfl_xor(s, 4, 32);
            s += __shfl_xor(s, 8, 32);
            rsum[v] = rsum[v] * c + s;
        }
#pragma unroll
        for (int d = 0; d < 10; ++d)
#pragma unroll
            for (int v = 0; v < 8; ++v) acc[d][v] *= corr[v];

        __syncthreads();
        // P tile -> LDS (row = M, col = kv), bf16
#pragma unroll
        for (int t2 = 0; t2 < 2; ++t2)
#pragma unroll
            for (int v = 0; v < 8; ++v)
                pT[(v + 8 * half) * ALD + t2 * 16 + ln] = (__bf16)sc[t2][v];

#if USE_TDM
        // drain our own LDS reads of the buffer being overwritten, then DMA
        asm volatile("s_wait_dscnt 0" ::: "memory");
        if (j + 1 < nsteps) {
            issueV(j + 1, cur ^ 1);
            __builtin_amdgcn_s_wait_tensorcnt(1);   // tile j complete
        } else {
            __builtin_amdgcn_s_wait_tensorcnt(0);   // final tile complete
        }
#else
        {   // manual staging: lane copies one V row with 16B vector moves
            const int kvr = kv0 + lane;
            __bf16* dst = &vS[cur][lane * HD_];
            if (kvr < L) {
                const __bf16* vp = Vb + base + (size_t)kvr * Dm;
                for (int c = 0; c < HD_; c += 8) *(v8bf*)(dst + c) = *(const v8bf*)(vp + c);
            } else {
                for (int c = 0; c < HD_; c += 8) *(v8bf*)(dst + c) = zero8bf();
            }
        }
#endif
        __syncthreads();

        // ---- O += P V ----
        const v16bf ap = ld_avec(&pT[ln * ALD], half);
#pragma unroll
        for (int d = 0; d < 10; ++d) {
            v16bf bv;
            const __bf16* col = &vS[cur][(16 * half) * HD_ + d * 16 + ln];
#pragma unroll
            for (int e = 0; e < 16; ++e) bv[e] = col[e * HD_];
            acc[d] = wmma_bf16(ap, bv, acc[d]);
        }
    }

    // ---- normalize + store bf16 ----
#pragma unroll
    for (int v = 0; v < 8; ++v) {
        const int m = v + 8 * half;
        const int row = qt * 16 + m;
        if (row < L) {
            const float inv = 1.0f / rsum[v];
            __bf16* op = Ob + base + (size_t)row * Dm;
#pragma unroll
            for (int d = 0; d < 10; ++d) op[d * 16 + ln] = (__bf16)(acc[d][v] * inv);
        }
    }
}

// ---------------------------------------------------------------------------
// alphas = sigmoid(h[:,:,D-1]) * mask; alphas_sum; resize by nt/sum
// ---------------------------------------------------------------------------
__global__ __launch_bounds__(256)
void k_alphas(const float* __restrict__ Hp, const float* __restrict__ AM,
              const int* __restrict__ NT, float* __restrict__ a_ws,
              float* __restrict__ a_out, float* __restrict__ sum_out, int Dm)
{
    __shared__ float red[256];
    const int b = blockIdx.x, tid = threadIdx.x;
    float s = 0.0f;
    for (int t = tid; t < T_; t += 256) {
        const float z = Hp[((size_t)b * T_ + t) * Dm + (Dm - 1)];
        const float a = (1.0f / (1.0f + __expf(-z))) * AM[(size_t)b * T_ + t];
        a_ws[(size_t)b * T_ + t] = a;
        s += a;
    }
    red[tid] = s; __syncthreads();
    for (int off = 128; off; off >>= 1) { if (tid < off) red[tid] += red[tid + off]; __syncthreads(); }
    const float total = red[0];
    if (tid == 0) sum_out[b] = total;
    const int nt = max(NT[b], 1);
    const float sc = (float)nt / total;
    for (int t = tid; t < T_; t += 256) {
        const float a = a_ws[(size_t)b * T_ + t] * sc;
        a_ws[(size_t)b * T_ + t] = a;
        a_out[(size_t)b * T_ + t] = a;
    }
}

// ---------------------------------------------------------------------------
// CIF scan: one wave; lanes 0..B-1 own batches; cross-batch any() via __any
// (faithful to the reference's scalar `active` coupling + do-while body).
// W must be pre-zeroed, shape [B, MT, T].
// ---------------------------------------------------------------------------
__global__ __launch_bounds__(32)
void k_cif(const float* __restrict__ alphas, const int* __restrict__ NT,
           float* __restrict__ W)
{
    const int lane = threadIdx.x;
    const bool on = lane < Bn_;
    const int nt = on ? max(NT[lane], 1) : 1;
    float integrate = 0.0f;
    int ti = 0;
    for (int t = 0; t < T_; ++t) {
        const float alpha = on ? alphas[(size_t)lane * T_ + t] : 0.0f;
        float aneed = 1.0f - integrate;
        integrate += alpha;
        float acur = alpha;
        bool ready = integrate >= 1.0f;
        bool active = true;
        for (int it = 0; it < 4; ++it) {   // MAX_INNER, first iter always runs
            if (active) {
                if (ready) integrate -= 1.0f;
                const float aint = ready ? aneed : acur;
                if (on) W[((size_t)lane * MT_ + ti) * T_ + t] = aint;  // overwrite
                const float rem = acur - aint;
                ti = min(ti + (ready ? 1 : 0), nt - 1);
                acur = rem;
                aneed = 1.0f;
                ready = integrate >= 1.0f;
            }
            active = __any((int)(on && ready)) != 0;
        }
        if (on && t < T_ - 1)
            W[((size_t)lane * MT_ + ti) * T_ + t] += acur;   // remainder scatter-add
    }
}

__global__ void k_fillf(float* p, float v, long n) {
    const long i = (long)blockIdx.x * blockDim.x + threadIdx.x;
    if (i < n) p[i] = v;
}

__global__ void k_cvt_bf16(const float* __restrict__ src, __bf16* __restrict__ dst, long n) {
    long i = (long)blockIdx.x * 256 + threadIdx.x;
    const long stride = (long)gridDim.x * 256;
    for (; i < n; i += stride) dst[i] = (__bf16)src[i];
}

__global__ void k_mask(const int* __restrict__ NT, int* __restrict__ out) {
    const int i = blockIdx.x * blockDim.x + threadIdx.x;
    if (i < Bn_ * MT_) {
        const int b = i / MT_, j = i % MT_;
        out[i] = (j < max(NT[b], 1)) ? 1 : 0;
    }
}

// ---------------------------------------------------------------------------
// Host orchestration
// ---------------------------------------------------------------------------
static void launch_gemm(hipStream_t st,
                        const void* A, int a_f32, long lda, long strideA,
                        const void* Bm, int b_f32, int b_kn, long ldb, long strideB,
                        const float* bias, const float* resid, long ldr, long strideR,
                        float* Cf, __bf16* Cb, long ldc, long strideC,
                        int M, int N, int K, int batch, float alpha, int act)
{
    dim3 g((unsigned)((N + 63) / 64), (unsigned)((M + 63) / 64), (unsigned)batch);
    k_gemm<<<g, dim3(128), 0, st>>>(A, a_f32, lda, strideA, Bm, b_f32, b_kn, ldb, strideB,
                                    bias, resid, ldr, strideR, Cf, Cb, ldc, strideC,
                                    M, N, K, alpha, act);
}

extern "C" void kernel_launch(void* const* d_in, const int* in_sizes, int n_in,
                              void* d_out, int out_size, void* d_ws, size_t ws_size,
                              hipStream_t stream)
{
    (void)in_sizes; (void)n_in; (void)out_size; (void)ws_size;

    const float* x_in  = (const float*)d_in[0];
    const float* amask = (const float*)d_in[1];
    const int*   ntok  = (const int*)d_in[2];
    enum { LN1S, LN1B, QW, QB, KW, VW, VB, OW, OB, LN2S, LN2B, F1W, F1B, F2W, F2B };
    const float* pre[15];  for (int i = 0; i < 15; ++i) pre[i]  = (const float*)d_in[3 + i];
    const float* post[15]; for (int i = 0; i < 15; ++i) post[i] = (const float*)d_in[18 + i];
    const float* cif_w  = (const float*)d_in[33];
    const float* cif_b  = (const float*)d_in[34];
    const float* proj_w = (const float*)d_in[35];
    const float* proj_b = (const float*)d_in[36];
    const float* lm_w   = (const float*)d_in[37];

    // output layout (tuple flattened): hid, out_mask, logits, alphas, alphas_sum
    float* out_hid = (float*)d_out;
    size_t off = (size_t)Bn_ * MT_ * OUT_;
    int*   out_mask_i = (int*)((float*)d_out + off);
    off += (size_t)Bn_ * MT_;
    float* out_logits = (float*)d_out + off;
    off += (size_t)Bn_ * MT_ * VOC_;
    float* out_alphas = (float*)d_out + off;
    off += (size_t)Bn_ * T_;
    float* out_asum = (float*)d_out + off;

    // workspace bump allocator
    char* wsp = (char*)d_ws;
    auto alloc = [&](size_t bytes) -> char* {
        char* r = wsp; wsp += (bytes + 255) & ~(size_t)255; return r;
    };
    const int Mpre  = Bn_ * T_;    // 6000
    const int Mpost = Bn_ * MT_;   // 600

    // weights -> bf16 once per launch (makes every big GEMM TDM-eligible)
    auto cvt = [&](const float* w, size_t n) -> __bf16* {
        __bf16* p = (__bf16*)alloc(n * 2);
        long blocks = (long)((n + 255) / 256); if (blocks > 2048) blocks = 2048;
        k_cvt_bf16<<<dim3((unsigned)blocks), dim3(256), 0, stream>>>(w, p, (long)n);
        return p;
    };
    const int widx[6] = { QW, KW, VW, OW, F1W, F2W };
    const size_t wsz[6] = { (size_t)D_ * D_, (size_t)D_ * D_, (size_t)D_ * D_,
                            (size_t)D_ * D_, (size_t)FFN_ * D_, (size_t)D_ * FFN_ };
    __bf16* pre_w[6];  for (int i = 0; i < 6; ++i) pre_w[i]  = cvt(pre[widx[i]],  wsz[i]);
    __bf16* post_w[6]; for (int i = 0; i < 6; ++i) post_w[i] = cvt(post[widx[i]], wsz[i]);
    __bf16* cifw_bf  = cvt(cif_w,  (size_t)D_ * (D_ - 1));
    __bf16* projw_bf = cvt(proj_w, (size_t)OUT_ * D_);
    __bf16* lmw_bf   = cvt(lm_w,   (size_t)VOC_ * D_);

    __bf16* hln   = (__bf16*)alloc((size_t)Mpre * D_ * 2);
    __bf16* qb    = (__bf16*)alloc((size_t)Mpre * D_ * 2);
    __bf16* kb    = (__bf16*)alloc((size_t)Mpre * D_ * 2);
    __bf16* vb    = (__bf16*)alloc((size_t)Mpre * D_ * 2);
    __bf16* ab    = (__bf16*)alloc((size_t)Mpre * D_ * 2);
    float*  x1    = (float*)alloc((size_t)Mpre * D_ * 4);
    __bf16* hln2  = (__bf16*)alloc((size_t)Mpre * D_ * 2);
    __bf16* fbuf  = (__bf16*)alloc((size_t)Mpre * FFN_ * 2);
    float*  hpre  = (float*)alloc((size_t)Mpre * D_ * 4);   // reused as hpost
    float*  alph  = (float*)alloc((size_t)Bn_ * T_ * 4);
    float*  wcif  = (float*)alloc((size_t)Bn_ * MT_ * T_ * 4);
    __bf16* cifb  = (__bf16*)alloc((size_t)Mpost * (D_ - 1) * 2);
    float*  h2    = (float*)alloc((size_t)Mpost * D_ * 4);
    __bf16* hpbf  = (__bf16*)alloc((size_t)Mpost * D_ * 2);

    const float qscale = 0.07905694150420949f;   // HD^-0.5 = 1/sqrt(160)

    auto encoder = [&](const float* xin, const float* const prm[15],
                       __bf16* const wbf[6], int Mrows, int L,
                       const int* vlenp, float* xout, __bf16* xout_bf) {
        k_layernorm<<<dim3((unsigned)Mrows), dim3(256), 0, stream>>>(xin, prm[LN1S], prm[LN1B], hln, D_);
        launch_gemm(stream, hln, 0, D_, 0, wbf[0], 0, 0, D_, 0, prm[QB], nullptr, 0, 0,
                    nullptr, qb, D_, 0, Mrows, D_, D_, 1, qscale, 0);
        launch_gemm(stream, hln, 0, D_, 0, wbf[1], 0, 0, D_, 0, nullptr, nullptr, 0, 0,
                    nullptr, kb, D_, 0, Mrows, D_, D_, 1, 1.0f, 0);
        launch_gemm(stream, hln, 0, D_, 0, wbf[2], 0, 0, D_, 0, prm[VB], nullptr, 0, 0,
                    nullptr, vb, D_, 0, Mrows, D_, D_, 1, 1.0f, 0);
        dim3 ga((unsigned)((L + 15) / 16), (unsigned)H_, (unsigned)Bn_);
        k_attn<<<ga, dim3(32), 0, stream>>>(qb, kb, vb, ab, L, D_, vlenp);
        launch_gemm(stream, ab, 0, D_, 0, wbf[3], 0, 0, D_, 0, prm[OB], xin, D_, 0,
                    x1, nullptr, D_, 0, Mrows, D_, D_, 1, 1.0f, 0);
        k_layernorm<<<dim3((unsigned)Mrows), dim3(256), 0, stream>>>(x1, prm[LN2S], prm[LN2B], hln2, D_);
        launch_gemm(stream, hln2, 0, D_, 0, wbf[4], 0, 0, D_, 0, prm[F1B], nullptr, 0, 0,
                    nullptr, fbuf, FFN_, 0, Mrows, FFN_, D_, 1, 1.0f, 1);
        launch_gemm(stream, fbuf, 0, FFN_, 0, wbf[5], 0, 0, FFN_, 0, prm[F2B], x1, D_, 0,
                    xout, xout_bf, D_, 0, Mrows, D_, FFN_, 1, 1.0f, 0);
    };

    // ---- pre encoder layer on [B*T, D] ----
    encoder(x_in, pre, pre_w, Mpre, T_, nullptr, hpre, nullptr);

    // ---- alphas / resize ----
    k_alphas<<<dim3((unsigned)Bn_), dim3(256), 0, stream>>>(hpre, amask, ntok, alph,
                                                            out_alphas, out_asum, D_);

    // ---- CIF scan -> weights [B, MT, T] ----
    const long wn = (long)Bn_ * MT_ * T_;
    k_fillf<<<dim3((unsigned)((wn + 255) / 256)), dim3(256), 0, stream>>>(wcif, 0.0f, wn);
    k_cif<<<dim3(1), dim3(32), 0, stream>>>(alph, ntok, wcif);

    // ---- cif = weights @ hidden[:, :, :D-1]  (batched; fallback path) ----
    launch_gemm(stream, wcif, 1, T_, (long)MT_ * T_,
                hpre, 1, 1, D_, (long)T_ * D_,
                nullptr, nullptr, 0, 0,
                nullptr, cifb, D_ - 1, (long)MT_ * (D_ - 1),
                MT_, D_ - 1, T_, Bn_, 1.0f, 0);

    // ---- h2 = cif @ cif_w.T + cif_b ----
    launch_gemm(stream, cifb, 0, D_ - 1, 0, cifw_bf, 0, 0, D_ - 1, 0, cif_b, nullptr, 0, 0,
                h2, nullptr, D_, 0, Mpost, D_, D_ - 1, 1, 1.0f, 0);

    // ---- post encoder layer on [B*MT, D] with length mask ----
    encoder(h2, post, post_w, Mpost, MT_, ntok, hpre /*reuse as hpost*/, hpbf);

    // ---- heads ----
    launch_gemm(stream, hpbf, 0, D_, 0, lmw_bf, 0, 0, D_, 0, nullptr, nullptr, 0, 0,
                out_logits, nullptr, VOC_, 0, Mpost, VOC_, D_, 1, 1.0f, 0);
    launch_gemm(stream, hpbf, 0, D_, 0, projw_bf, 0, 0, D_, 0, proj_b, nullptr, 0, 0,
                out_hid, nullptr, OUT_, 0, Mpost, OUT_, D_, 1, 1.0f, 0);

    k_mask<<<dim3((unsigned)((Bn_ * MT_ + 255) / 256)), dim3(256), 0, stream>>>(ntok, out_mask_i);
}
